// HybridLayer_90769838834351
// MI455X (gfx1250) — matrix operand order repaired and checked by reading, not compiled, not run on hardware
//
#include <hip/hip_runtime.h>

typedef __attribute__((ext_vector_type(16))) _Float16 v16h;
typedef __attribute__((ext_vector_type(8)))  float    v8f;
typedef __attribute__((ext_vector_type(4)))  float    v4f;

#define N_BATCH 128
#define C_IN    32
#define T_LEN   8192
#define N_FILT  17
#define FRAME   96    // unified tap frame: columns 0..95 <-> r = c-47 in [-47,48]
#define HALO_L  47
#define WIN     224   // s-window halves per block (needs 223)

// ---------------- Pass 1: channel reduction s[n,t] = sum_c x[n,c,t] ----------
// x is 134 MB streamed exactly once -> non-temporal loads (don't pollute L2);
// s is 4 MB written once, re-read by pass 2 -> regular (RT) stores keep it hot.
__global__ __launch_bounds__(256)
void chansum_kernel(const float* __restrict__ x, float* __restrict__ s) {
    int tid = blockIdx.x * 256 + threadIdx.x;       // 262144 threads total
    int n  = tid >> 11;                              // / (8192/4)
    int tq = tid & 2047;                             // float4 index along t
    const v4f* xb = (const v4f*)(x + (size_t)n * C_IN * T_LEN) + tq;
    v4f acc = {0.f, 0.f, 0.f, 0.f};
#pragma unroll
    for (int c = 0; c < C_IN; ++c) {
        v4f v = __builtin_nontemporal_load(&xb[c * (T_LEN / 4)]);
        acc += v;
    }
    ((v4f*)(s + (size_t)n * T_LEN))[tq] = acc;
}

// ---------------- Filter-weight generator (exact in f16) ---------------------
// Row r of the 32x96 W matrix: r in [0,15] -> filters 0..15, r==16 -> filter 16.
__device__ __forceinline__ float filt_weight(int r, int c) {
    if (r > 16) return 0.0f;
    int f = r;
    int len, qs = 0, fam;
    if (f < 6)       { fam = 0; len = 2 << f; }          // alternating, -1 on even j
    else if (f < 12) { fam = 1; len = 2 << (f - 6); }    // alternating, -1 on odd j
    else             { fam = 2; qs = f - 12; len = 6 << qs; } // ramp family, ext = 6q
    int pad = (len - 1) >> 1;                            // XLA SAME, stride 1
    int j = c - HALO_L + pad;
    if (j < 0 || j >= len) return 0.0f;
    if (fam == 0) return (j & 1) ? 1.0f : -1.0f;
    if (fam == 1) return (j & 1) ? -1.0f : 1.0f;
    int q   = 1 << qs;
    int seg = j >> qs;
    int pos = j & (q - 1);
    float base = (seg & 1) ? (float)(q - pos) * (1.0f / (float)q)
                           : (float)(pos + 1) * (1.0f / (float)q);
    float v = base * base;                               // exact: (int<=256)/2^k
    float m = (seg == 2 || seg == 3) ? 2.0f : -1.0f;
    return m * v;
}

// ---------------- Pass 2: 17-filter conv as WMMA matmul + ReLU ---------------
// Block: 256 threads = 8 waves; block covers 128 t positions of one batch row.
// Wave w computes Y(17 filters x 16 t) via v_wmma_f32_16x16x32_f16, split-f16:
// s = s_hi + s_lo (two f16), weights exact in f16, f32 accumulate -> ~f32 acc.
__global__ __launch_bounds__(256)
void conv_wmma_kernel(const float* __restrict__ s, float* __restrict__ out) {
    __shared__ _Float16 Wl[32][FRAME];   // 6 KB: filter matrix (f16-exact)
    __shared__ _Float16 sh[WIN];         // hi f16 of s window
    __shared__ _Float16 sl[WIN];         // lo f16 residual

    const int n   = blockIdx.y;
    const int T0  = blockIdx.x * 128;
    const int tid = threadIdx.x;

    // Stage W (uniform trip count -> no divergence)
    for (int i = tid; i < 32 * FRAME; i += 256) {
        int r = i / FRAME, c = i - r * FRAME;
        Wl[r][c] = (_Float16)filt_weight(r, c);
    }
    // Stage s window [T0-47, T0+176] with hi/lo split and zero padding
    for (int i = tid; i < WIN; i += 256) {
        int t = T0 - HALO_L + i;
        float v = (t >= 0 && t < T_LEN) ? s[(size_t)n * T_LEN + t] : 0.0f;
        _Float16 h = (_Float16)v;
        sh[i] = h;
        sl[i] = (_Float16)(v - (float)h);
    }
    __syncthreads();

    const int wave   = tid >> 5;
    const int lane   = tid & 31;
    const int nIdx   = lane & 15;   // == A-row m and D column N
    const int hiHalf = lane >> 4;   // upper half-wave?

    v8f acc0 = {};  // filters 0..15
    v8f acc1 = {};  // filter 16 (rows 16..31 of W, mostly zero)

#pragma unroll
    for (int kk = 0; kk < FRAME; kk += 32) {
        // A operand layout: contiguous halves, matching B
        //   K = hh + (upper? 16 : 0)
        v16h a0, a1;
#pragma unroll
        for (int hh = 0; hh < 16; ++hh) {
            int K = hh + (hiHalf ? 16 : 0);
            a0[hh] = Wl[nIdx][kk + K];
            a1[hh] = Wl[16 + nIdx][kk + K];
        }
        // B operand: B[k', n] = s(T0 + 16*wave + n + kk + k' - 47)
        //   lanes 0-15: K = hh ; lanes 16-31: K = hh + 16  -> contiguous halves
        int ib = wave * 16 + nIdx + kk + (hiHalf ? 16 : 0);
        v16h bh, bl;
#pragma unroll
        for (int hh = 0; hh < 16; ++hh) { bh[hh] = sh[ib + hh]; bl[hh] = sl[ib + hh]; }

        acc0 = __builtin_amdgcn_wmma_f32_16x16x32_f16(false, a0, false, bh, (short)0, acc0, false, false);
        acc0 = __builtin_amdgcn_wmma_f32_16x16x32_f16(false, a0, false, bl, (short)0, acc0, false, false);
        acc1 = __builtin_amdgcn_wmma_f32_16x16x32_f16(false, a1, false, bh, (short)0, acc1, false, false);
        acc1 = __builtin_amdgcn_wmma_f32_16x16x32_f16(false, a1, false, bl, (short)0, acc1, false, false);
    }

    // D layout: VGPR i -> M = i + 8*hiHalf, N = lane%16. ReLU + NT store
    // (output is write-once, never re-read: keep it out of L2).
    int t = T0 + wave * 16 + nIdx;
    size_t outBase = (size_t)n * N_FILT * T_LEN + t;
#pragma unroll
    for (int i = 0; i < 8; ++i) {
        int f = i + (hiHalf ? 8 : 0);
        float v0 = acc0[i];
        v0 = v0 > 0.0f ? v0 : 0.0f;
        __builtin_nontemporal_store(v0, &out[outBase + (size_t)f * T_LEN]);
    }
    if (hiHalf == 0) {  // filter 16 lives in acc1 row M=0 only
        float v1 = acc1[0];
        v1 = v1 > 0.0f ? v1 : 0.0f;
        __builtin_nontemporal_store(v1, &out[outBase + (size_t)16 * T_LEN]);
    }
}

extern "C" void kernel_launch(void* const* d_in, const int* in_sizes, int n_in,
                              void* d_out, int out_size, void* d_ws, size_t ws_size,
                              hipStream_t stream) {
    const float* x = (const float*)d_in[0];
    float* out = (float*)d_out;
    float* s   = (float*)d_ws;   // needs 128*8192*4 = 4 MB scratch

    chansum_kernel<<<dim3((N_BATCH * T_LEN / 4) / 256), dim3(256), 0, stream>>>(x, s);
    conv_wmma_kernel<<<dim3(T_LEN / 128, N_BATCH), dim3(256), 0, stream>>>(s, out);
}
